// PD_LTS_Heavy_49572512530926
// MI455X (gfx1250) — matrix operand order repaired and checked by reading, not compiled, not run on hardware
//
#include <hip/hip_runtime.h>
#include <hip/hip_bf16.h>

typedef __attribute__((ext_vector_type(16))) _Float16 v16h;
typedef __attribute__((ext_vector_type(8)))  float    v8f;

#define B_   8
#define N_   2048
#define K_   32
#define H_   64
#define O_   128
#define S_   (B_*N_*K_)   /* 524288 samples */
#define P_   (B_*N_)      /* 16384 points  */
#define EPS_   1e-5f
#define SLOPE_ 0.05f

/* workspace layout (float offsets) */
#define X_OFF   0                      /* x[S][6]            */
#define ST1_OFF (S_*6)                 /* 42: sum[6]+xx[36]  */
#define W1P_OFF (ST1_OFF+48)           /* 384 folded W1      */
#define B1P_OFF (W1P_OFF+384)          /* 64 folded bias1    */
#define ST2_OFF (B1P_OFF+64)           /* 256: sum/sumsq z   */
#define A2_OFF  (ST2_OFF+256)          /* 128                */
#define C2_OFF  (A2_OFF+128)           /* 128                */
#define WS_PARAM_FLOATS (48+384+64+256+128+128)

__device__ __forceinline__ float wsum32(float v){
  #pragma unroll
  for (int o = 16; o; o >>= 1) v += __shfl_xor(v, o, 32);
  return v;
}

/* ------------------------------------------------------------------ */
/* K1: per-point KNN (top-32 by squared distance), emit x = [f, fn-f], */
/* accumulate global mean / second-moment of x for analytic BN1.       */
/* ------------------------------------------------------------------ */
__global__ __launch_bounds__(128) void knn_x_stats(const float* __restrict__ f,
                                                   float* __restrict__ ws){
  __shared__ float fs[N_*3];        /* 24 KB: this batch's points */
  __shared__ float dist[4][N_];     /* 32 KB: one row per wave    */
  __shared__ int   sel[4][K_];
  __shared__ float sacc[42];
  const int t    = threadIdx.x;
  const int b    = blockIdx.x >> 9;            /* 512 blocks per batch */
  const int row0 = (blockIdx.x & 511) * 4;

  for (int i = t; i < N_*3; i += 128) fs[i] = f[(size_t)b*N_*3 + i];
  if (t < 42) sacc[t] = 0.f;
  __syncthreads();

  const int w = t >> 5, lane = t & 31;
  const int m = row0 + w;
  const float fmx = fs[m*3+0], fmy = fs[m*3+1], fmz = fs[m*3+2];

  #pragma unroll 4
  for (int i = 0; i < N_/32; ++i){
    int n = lane + 32*i;
    float dx = fs[n*3+0]-fmx, dy = fs[n*3+1]-fmy, dz = fs[n*3+2]-fmz;
    dist[w][n] = dx*dx + dy*dy + dz*dz;
  }
  __builtin_amdgcn_wave_barrier();

  /* 32 rounds of wave-wide argmin over the LDS distance row */
  for (int kk = 0; kk < K_; ++kk){
    float dmin = 3.4e38f; int imin = N_;
    for (int i = 0; i < N_/32; ++i){
      int n = lane + 32*i;
      float d = dist[w][n];
      if (d < dmin || (d == dmin && n < imin)){ dmin = d; imin = n; }
    }
    #pragma unroll
    for (int off = 16; off; off >>= 1){
      float od = __shfl_xor(dmin, off, 32);
      int   oi = __shfl_xor(imin, off, 32);
      if (od < dmin || (od == dmin && oi < imin)){ dmin = od; imin = oi; }
    }
    if (lane == 0){ sel[w][kk] = imin; dist[w][imin] = 3.4e38f; }
    __builtin_amdgcn_wave_barrier();
  }

  /* lane k -> neighbor k: build x and moments */
  const int n = sel[w][lane];
  float xv[6];
  xv[0]=fmx; xv[1]=fmy; xv[2]=fmz;
  xv[3]=fs[n*3+0]-fmx; xv[4]=fs[n*3+1]-fmy; xv[5]=fs[n*3+2]-fmz;
  size_t sb = ((size_t)(b*N_ + m)*K_ + lane)*6;
  #pragma unroll
  for (int c = 0; c < 6; ++c) ws[X_OFF + sb + c] = xv[c];
  #pragma unroll
  for (int c = 0; c < 6; ++c){
    float v = wsum32(xv[c]);
    if (lane == 0) atomicAdd(&sacc[c], v);
  }
  #pragma unroll
  for (int c = 0; c < 6; ++c){
    #pragma unroll
    for (int d = 0; d < 6; ++d){
      float v = wsum32(xv[c]*xv[d]);
      if (lane == 0) atomicAdd(&sacc[6 + c*6 + d], v);
    }
  }
  __syncthreads();
  if (t < 42) atomicAdd(&ws[ST1_OFF + t], sacc[t]);
}

/* ------------------------------------------------------------------ */
/* K2: fold BN1 analytically: var of affine layer from x moments.      */
/* ------------------------------------------------------------------ */
__global__ void fold_bn1(const float* __restrict__ W1, const float* __restrict__ b1,
                         const float* __restrict__ g1, const float* __restrict__ be1,
                         float* __restrict__ ws){
  const int h = threadIdx.x;
  if (h >= H_) return;
  const float inv = 1.0f / (float)S_;
  float mu = b1[h];
  #pragma unroll
  for (int c = 0; c < 6; ++c) mu += W1[h*6+c] * (ws[ST1_OFF+c]*inv);
  float e2 = 0.f;
  #pragma unroll
  for (int c = 0; c < 6; ++c)
    #pragma unroll
    for (int d = 0; d < 6; ++d)
      e2 += W1[h*6+c]*W1[h*6+d]*(ws[ST1_OFF+6+c*6+d]*inv);
  e2 += 2.f*b1[h]*(mu - b1[h]) + b1[h]*b1[h];
  float var = e2 - mu*mu;
  float a = g1[h] * rsqrtf(var + EPS_);
  #pragma unroll
  for (int c = 0; c < 6; ++c) ws[W1P_OFF + h*6 + c] = W1[h*6+c]*a;
  ws[B1P_OFF + h] = (b1[h] - mu)*a + be1[h];
}

/* ---- shared fragment-builder helpers (ISA 7.12.2 layouts) ---------- */
union FragH { v16h v; _Float16 e[16]; };
union FragF { v8f  v; float    e[8];  };

/* ------------------------------------------------------------------ */
/* K3: h1 = lrelu(x@W1' + b1'); WMMA GEMM z = h1@W2^T; accumulate      */
/* per-channel sum(z), sum(z^2) for BN2 (raw, bias folded later).      */
/* ------------------------------------------------------------------ */
__global__ __launch_bounds__(256) void mm2_stats(const float* __restrict__ W2,
                                                 float* __restrict__ ws){
  __shared__ _Float16 h1s[128][72];     /* 18 KB, padded rows */
  __shared__ _Float16 w2s[128][64];     /* 16 KB              */
  __shared__ float    w1ps[448];
  __shared__ float    acc[256];
  const int t = threadIdx.x;
  for (int i = t; i < 448; i += 256) w1ps[i] = ws[W1P_OFF + i];
  for (int i = t; i < O_*H_; i += 256) w2s[i>>6][i&63] = (_Float16)W2[i];
  acc[t] = 0.f;
  __syncthreads();

  { /* build h1 tile: 2 threads per sample, 32 channels each */
    const int sl = t >> 1, c0 = (t & 1) * 32;
    const size_t sg = (size_t)blockIdx.x*128 + sl;
    float xv[6];
    #pragma unroll
    for (int j = 0; j < 6; ++j) xv[j] = ws[X_OFF + sg*6 + j];
    for (int c = c0; c < c0+32; ++c){
      float h = w1ps[384 + c];
      #pragma unroll
      for (int j = 0; j < 6; ++j) h += xv[j]*w1ps[c*6 + j];
      h = h > 0.f ? h : SLOPE_*h;
      h1s[sl][c] = (_Float16)h;
    }
  }
  __syncthreads();

  const int wv = t >> 5, lane = t & 31, g = lane >> 4, ml = lane & 15;
  const int sb = wv * 16;
  FragH a0, a1, b0, b1;
  #pragma unroll
  for (int j = 0; j < 16; ++j){
    int k = 16*g + j;        /* A 16x32 f16 layout */
    a0.e[j] = h1s[sb+ml][k];
    a1.e[j] = h1s[sb+ml][32 + k];
  }
  for (int ot = 0; ot < 8; ++ot){
    const int o = ot*16 + ml;
    #pragma unroll
    for (int e = 0; e < 16; ++e){             /* B 32x16 f16 layout */
      b0.e[e] = w2s[o][16*g + e];
      b1.e[e] = w2s[o][32 + 16*g + e];
    }
    FragF c; v8f z = {}; c.v = z;
    c.v = __builtin_amdgcn_wmma_f32_16x16x32_f16(false, a0.v, false, b0.v, (short)0, c.v, false, false);
    c.v = __builtin_amdgcn_wmma_f32_16x16x32_f16(false, a1.v, false, b1.v, (short)0, c.v, false, false);
    float s = 0.f, q = 0.f;
    #pragma unroll
    for (int r = 0; r < 8; ++r){ s += c.e[r]; q += c.e[r]*c.e[r]; }
    s += __shfl_xor(s, 16, 32);
    q += __shfl_xor(q, 16, 32);
    if (lane < 16){ atomicAdd(&acc[o], s); atomicAdd(&acc[128+o], q); }
  }
  __syncthreads();
  atomicAdd(&ws[ST2_OFF + t], acc[t]);
}

/* ------------------------------------------------------------------ */
/* K4: fold BN2 + bias2 into per-channel scale/shift (y = a2*zraw+c2). */
/* ------------------------------------------------------------------ */
__global__ void fold_bn2(const float* __restrict__ b2, const float* __restrict__ g2,
                         const float* __restrict__ be2, float* __restrict__ ws){
  const int o = threadIdx.x;
  if (o >= O_) return;
  const float inv = 1.0f / (float)S_;
  float mraw = ws[ST2_OFF + o] * inv;
  float q    = ws[ST2_OFF + 128 + o] * inv;
  float mean = mraw + b2[o];
  float e2   = q + 2.f*b2[o]*mraw + b2[o]*b2[o];
  float var  = e2 - mean*mean;
  float a    = g2[o] * rsqrtf(var + EPS_);
  ws[A2_OFF + o] = a;
  ws[C2_OFF + o] = a*b2[o] + be2[o] - mean*a;
}

/* ------------------------------------------------------------------ */
/* K5: recompute h1, WMMA GEMM, BN2+LeakyReLU, max over 32 neighbors,  */
/* write out[P][131] = [max-features(128), f(3)].                      */
/* One wave owns one point (32 neighbors = 2 M-blocks).                */
/* ------------------------------------------------------------------ */
__global__ __launch_bounds__(256) void mm2_out(const float* __restrict__ W2,
                                               const float* __restrict__ f,
                                               float* __restrict__ ws,
                                               float* __restrict__ out){
  __shared__ _Float16 h1s[256][72];     /* 36 KB */
  __shared__ _Float16 w2s[128][64];     /* 16 KB */
  __shared__ float    w1ps[448];
  __shared__ float    a2s[128], c2s[128];
  const int t = threadIdx.x;
  for (int i = t; i < 448; i += 256) w1ps[i] = ws[W1P_OFF + i];
  for (int i = t; i < O_*H_; i += 256) w2s[i>>6][i&63] = (_Float16)W2[i];
  if (t < 128){ a2s[t] = ws[A2_OFF + t]; c2s[t] = ws[C2_OFF + t]; }
  __syncthreads();

  { /* each thread builds all 64 channels of one sample */
    const size_t sg = (size_t)blockIdx.x*256 + t;
    float xv[6];
    #pragma unroll
    for (int j = 0; j < 6; ++j) xv[j] = ws[X_OFF + sg*6 + j];
    for (int c = 0; c < 64; ++c){
      float h = w1ps[384 + c];
      #pragma unroll
      for (int j = 0; j < 6; ++j) h += xv[j]*w1ps[c*6 + j];
      h = h > 0.f ? h : SLOPE_*h;
      h1s[t][c] = (_Float16)h;
    }
  }
  __syncthreads();

  const int wv = t >> 5, lane = t & 31, g = lane >> 4, ml = lane & 15;
  const int p   = blockIdx.x*8 + wv;     /* global point */
  const int sb0 = wv * 32;               /* this point's 32 neighbors in LDS */
  FragH a00, a01, a10, a11, b0, b1;
  #pragma unroll
  for (int j = 0; j < 16; ++j){
    int k = 16*g + j;
    a00.e[j] = h1s[sb0      + ml][k];
    a01.e[j] = h1s[sb0      + ml][32 + k];
    a10.e[j] = h1s[sb0 + 16 + ml][k];
    a11.e[j] = h1s[sb0 + 16 + ml][32 + k];
  }
  float mx[8];
  for (int ot = 0; ot < 8; ++ot){
    const int o = ot*16 + ml;
    #pragma unroll
    for (int e = 0; e < 16; ++e){
      b0.e[e] = w2s[o][16*g + e];
      b1.e[e] = w2s[o][32 + 16*g + e];
    }
    const float ya = a2s[o], yc = c2s[o];
    float mloc = -3.4e38f;
    {
      FragF c; v8f z = {}; c.v = z;
      c.v = __builtin_amdgcn_wmma_f32_16x16x32_f16(false, a00.v, false, b0.v, (short)0, c.v, false, false);
      c.v = __builtin_amdgcn_wmma_f32_16x16x32_f16(false, a01.v, false, b1.v, (short)0, c.v, false, false);
      #pragma unroll
      for (int r = 0; r < 8; ++r){
        float y = ya*c.e[r] + yc;
        y = y > 0.f ? y : SLOPE_*y;
        mloc = fmaxf(mloc, y);
      }
    }
    {
      FragF c; v8f z = {}; c.v = z;
      c.v = __builtin_amdgcn_wmma_f32_16x16x32_f16(false, a10.v, false, b0.v, (short)0, c.v, false, false);
      c.v = __builtin_amdgcn_wmma_f32_16x16x32_f16(false, a11.v, false, b1.v, (short)0, c.v, false, false);
      #pragma unroll
      for (int r = 0; r < 8; ++r){
        float y = ya*c.e[r] + yc;
        y = y > 0.f ? y : SLOPE_*y;
        mloc = fmaxf(mloc, y);
      }
    }
    mloc = fmaxf(mloc, __shfl_xor(mloc, 16, 32));
    mx[ot] = mloc;
  }
  if (lane < 16){
    const size_t ob = (size_t)p * 131;
    #pragma unroll
    for (int ot = 0; ot < 8; ++ot) out[ob + ot*16 + lane] = mx[ot];
    if (lane < 3) out[ob + 128 + lane] = f[(size_t)p*3 + lane];
  }
}

/* ------------------------------------------------------------------ */
extern "C" void kernel_launch(void* const* d_in, const int* in_sizes, int n_in,
                              void* d_out, int out_size, void* d_ws, size_t ws_size,
                              hipStream_t stream){
  (void)in_sizes; (void)out_size;
  if (n_in < 9) return;
  const float* f   = (const float*)d_in[0];
  const float* W1  = (const float*)d_in[1];
  const float* b1  = (const float*)d_in[2];
  const float* g1  = (const float*)d_in[3];
  const float* be1 = (const float*)d_in[4];
  const float* W2  = (const float*)d_in[5];
  const float* b2  = (const float*)d_in[6];
  const float* g2  = (const float*)d_in[7];
  const float* be2 = (const float*)d_in[8];
  float* out = (float*)d_out;
  float* ws  = (float*)d_ws;
  if (ws_size < (size_t)(ST1_OFF + WS_PARAM_FLOATS) * sizeof(float)) return;

  /* zero the stats/param tail each call (deterministic start) */
  hipMemsetAsync(ws + ST1_OFF, 0, WS_PARAM_FLOATS*sizeof(float), stream);

  hipLaunchKernelGGL(knn_x_stats, dim3(B_*(N_/4)), dim3(128), 0, stream, f, ws);
  hipLaunchKernelGGL(fold_bn1,    dim3(1),         dim3(64),  0, stream, W1, b1, g1, be1, ws);
  hipLaunchKernelGGL(mm2_stats,   dim3(S_/128),    dim3(256), 0, stream, W2, ws);
  hipLaunchKernelGGL(fold_bn2,    dim3(1),         dim3(128), 0, stream, b2, g2, be2, ws);
  hipLaunchKernelGGL(mm2_out,     dim3(P_/8),      dim3(256), 0, stream, W2, f, ws, out);
}
